// Model_74603581931870
// MI455X (gfx1250) — matrix operand order repaired and checked by reading, not compiled, not run on hardware
//
#include <hip/hip_runtime.h>
#include <hip/hip_bf16.h>

// ---------------- types for WMMA ----------------
typedef __attribute__((ext_vector_type(16))) __bf16 v16bf;
typedef __attribute__((ext_vector_type(8)))  float  v8f;

#define CIN   256
#define MED   128
#define HW    256
#define IMGPIX 65536          // 256*256
#define NPIX  262144          // 4*65536
#define AN    9
#define NSCORE 2359296        // NPIX*AN
#define TOPK  3000
#define CANDMAX 6144
#define SORTN 8192
#define STRB  40              // strip row stride (bf16 elems): 80B = 16B-aligned, bank-friendly
#define WROW  40              // weight-tile row stride (bf16 elems)

__device__ __forceinline__ unsigned short f32_to_bf16(float f) {
    unsigned int u = __float_as_uint(f);
    unsigned int lsb = (u >> 16) & 1u;
    u += 0x7FFFu + lsb;                 // round to nearest even
    return (unsigned short)(u >> 16);
}

__device__ __forceinline__ unsigned int fkey(float f) {
    unsigned int u = __float_as_uint(f);
    return (u & 0x80000000u) ? ~u : (u | 0x80000000u); // monotone: bigger float -> bigger key
}

// A-fragment (16x32 bf16): lane half picks K-origin 0/8; quads at +0 and +16 elems
__device__ __forceinline__ v16bf load_frag_a(const unsigned short* base) {
    union { v16bf v; uint4 q[2]; } u;
    u.q[0] = *(const uint4*)(base);
    u.q[1] = *(const uint4*)(base + 16);
    return u.v;
}
// B-fragment (32x16 bf16): lane half picks K-origin 0/8; quads at +0 and +16 elems
__device__ __forceinline__ v16bf load_frag_b(const unsigned short* base) {
    union { v16bf v; uint4 q[2]; } u;
    u.q[0] = *(const uint4*)(base);
    u.q[1] = *(const uint4*)(base + 16);
    return u.v;
}

// =====================================================================
// Kernel 0: repack prepare_w [128,256,3,3] f32 -> bf16 [t][cc][oc][32c]
// =====================================================================
__global__ __launch_bounds__(256) void repack_weights(
    const float* __restrict__ w, unsigned short* __restrict__ wpk)
{
    const int i = blockIdx.x * 256 + threadIdx.x;       // over 9*256*128
    if (i >= 9 * 256 * MED) return;
    const int c  = i & 31;
    const int oc = (i >> 5) & 127;
    const int cc = (i >> 12) & 7;
    const int t  = i >> 15;
    wpk[i] = f32_to_bf16(w[((size_t)oc * CIN + cc * 32 + c) * 9 + t]);
}

// =====================================================================
// Kernel 1: 3x3 conv, implicit GEMM, bf16 WMMA, f32 accumulate.
// WG = 256 thr (8 waves). Tile: 128 oc x 128 px (one row segment).
// Wave = 32 oc x 64 px -> 8 accumulators, 8 WMMAs per (ky,cc,kx) step.
// Halo strip (130 px x 32 ch) staged once per (ky,cc), reused for 3 kx taps.
// =====================================================================
__global__ __launch_bounds__(256) void conv3x3_wmma(
    const float* __restrict__ x,
    const unsigned short* __restrict__ wpk,
    const float* __restrict__ bias,
    unsigned short* __restrict__ feat)
{
    __shared__ __align__(16) unsigned short wlds[3 * MED * WROW]; // 30 KB
    __shared__ __align__(16) unsigned short strip[130 * STRB];    // 10.2 KB

    const int tid  = threadIdx.x;
    const int wave = tid >> 5;
    const int lane = tid & 31;
    const int mquad = wave & 3;     // oc group base = mquad*32
    const int nhalf = wave >> 2;    // px base = nhalf*64

    const int p0  = blockIdx.x * 128;
    const int b   = p0 >> 16;
    const int rem = p0 & (IMGPIX - 1);
    const int y   = rem >> 8;
    const int x0  = rem & 255;

    const int mrow = lane & 15;
    const int ncol = lane & 15;
    const int hi   = lane >> 4;
    const int kbA  = hi ? 8 : 0;
    const int kbB  = hi ? 8 : 0;

    v8f acc[2][4];
    #pragma unroll
    for (int i = 0; i < 2; ++i)
        #pragma unroll
        for (int j = 0; j < 4; ++j)
            #pragma unroll
            for (int r = 0; r < 8; ++r) acc[i][j][r] = 0.0f;

    for (int ky = 0; ky < 3; ++ky) {
        const int yy = y + ky - 1;
        const bool yok = (unsigned)yy < 256u;
        for (int cc = 0; cc < 8; ++cc) {
            // ---- stage weights (3 kx taps) via uint4 copies of repacked bf16 ----
            for (int q = tid; q < 1536; q += 256) {
                const int kx = q >> 9;
                const int r  = q & 511;
                const int oc = r >> 2;
                const int qi = r & 3;
                const uint4 v = ((const uint4*)wpk)[((((ky * 3 + kx) * 8 + cc) * MED + oc) << 2) + qi];
                *(uint4*)(wlds + (kx * MED + oc) * WROW + qi * 8) = v;
            }
            // ---- stage halo strip: 130 px x 32 ch, zero-padded borders ----
            for (int e = tid; e < 130 * 32; e += 256) {
                const int c  = e / 130;
                const int ph = e - c * 130;
                const int xx = x0 - 1 + ph;
                float v = 0.f;
                if (yok && (unsigned)xx < 256u) {
                    const float* xp = x + (((size_t)b * CIN + (cc * 32 + c)) * 256 + yy) * 256 + xx;
                    v = *xp;
                    if (cc < 7) __builtin_prefetch(xp + 32 * IMGPIX, 0, 1);
                }
                strip[ph * STRB + c] = f32_to_bf16(v);
            }
            __syncthreads();

            // ---- 3 kx taps from the same strip (address shift), 8 WMMAs each ----
            #pragma unroll
            for (int kx = 0; kx < 3; ++kx) {
                const unsigned short* wbase =
                    wlds + (kx * MED + mquad * 32 + mrow) * WROW + kbA;
                const v16bf a0 = load_frag_a(wbase);
                const v16bf a1 = load_frag_a(wbase + 16 * WROW);
                #pragma unroll
                for (int j = 0; j < 4; ++j) {
                    const int n = nhalf * 64 + j * 16 + ncol;   // px offset in tile
                    const v16bf bf = load_frag_b(strip + (n + kx) * STRB + kbB);
                    acc[0][j] = __builtin_amdgcn_wmma_f32_16x16x32_bf16(
                        false, a0, false, bf, (short)0, acc[0][j], false, false);
                    acc[1][j] = __builtin_amdgcn_wmma_f32_16x16x32_bf16(
                        false, a1, false, bf, (short)0, acc[1][j], false, false);
                }
            }
            __syncthreads();
        }
    }

    // ---- store: D layout -> feat bf16 [px][oc], packed pair stores ----
    #pragma unroll
    for (int i = 0; i < 2; ++i) {
        const int ocb = mquad * 32 + i * 16 + hi * 8;
        #pragma unroll
        for (int j = 0; j < 4; ++j) {
            const int px = p0 + nhalf * 64 + j * 16 + ncol;
            unsigned int* dst = (unsigned int*)(feat + (size_t)px * MED + ocb);
            #pragma unroll
            for (int r = 0; r < 4; ++r) {
                const unsigned int lo = f32_to_bf16(acc[i][j][2 * r]     + bias[ocb + 2 * r]);
                const unsigned int hh = f32_to_bf16(acc[i][j][2 * r + 1] + bias[ocb + 2 * r + 1]);
                dst[r] = lo | (hh << 16);
            }
        }
    }
}

// =====================================================================
// Kernel 2: fused 1x1 heads (cls+reg as one 64x128 GEMM) + box decode.
// =====================================================================
__global__ __launch_bounds__(256) void heads_decode_wmma(
    const unsigned short* __restrict__ feat,
    const float* __restrict__ cls_w, const float* __restrict__ cls_b,
    const float* __restrict__ reg_w, const float* __restrict__ reg_b,
    float* __restrict__ score)
{
    __shared__ __align__(16) unsigned short wlds[64 * 128]; // 16KB combined weights
    __shared__ __align__(16) unsigned short blds[32 * 128]; // 8KB  feat tile
    __shared__ float dlds[64 * 33];
    __shared__ float bias_s[64];

    const int tid  = threadIdx.x;
    const int wave = tid >> 5;
    const int lane = tid & 31;
    const int p0   = blockIdx.x * 32;

    for (int i = tid; i < 64 * 128; i += 256) {
        const int row = i >> 7, c = i & 127;
        float v = 0.f;
        if (row < 18)      v = cls_w[row * 128 + c];
        else if (row < 54) v = reg_w[(row - 18) * 128 + c];
        wlds[i] = f32_to_bf16(v);
    }
    if (tid < 64) {
        float v = 0.f;
        if (tid < 18)      v = cls_b[tid];
        else if (tid < 54) v = reg_b[tid - 18];
        bias_s[tid] = v;
    }
    {
        const uint4* src = (const uint4*)(feat + (size_t)p0 * 128);
        uint4* dst = (uint4*)blds;
        for (int i = tid; i < 512; i += 256) dst[i] = src[i];
    }
    __syncthreads();

    const int mtile = wave & 3, ntile = wave >> 2;
    const int mrow = lane & 15, ncol = lane & 15, hi = lane >> 4;
    const int kbA = hi ? 8 : 0;
    const int kbB = hi ? 8 : 0;

    v8f acc;
    #pragma unroll
    for (int i = 0; i < 8; ++i) acc[i] = 0.0f;

    #pragma unroll
    for (int kc = 0; kc < 4; ++kc) {
        const int kb = kc * 32;
        const v16bf a = load_frag_a(wlds + (mtile * 16 + mrow) * 128 + kb + kbA);
        const v16bf bf = load_frag_b(blds + (ntile * 16 + ncol) * 128 + kb + kbB);
        acc = __builtin_amdgcn_wmma_f32_16x16x32_bf16(
            false, a, false, bf, (short)0, acc, false, false);
    }

    #pragma unroll
    for (int r = 0; r < 8; ++r) {
        const int m = mtile * 16 + r + hi * 8;
        const int n = ntile * 16 + ncol;
        dlds[m * 33 + n] = acc[r] + bias_s[m];
    }
    __syncthreads();

    for (int i = tid; i < 32 * AN; i += 256) {
        const int pl = i / AN, a = i - pl * AN;
        const int p  = p0 + pl;
        const int rem = p & (IMGPIX - 1);
        const float fy = (float)(rem >> 8);
        const float fx = (float)(rem & 255);
        const float c0 = dlds[(2 * a) * 33 + pl];
        const float c1 = dlds[(2 * a + 1) * 33 + pl];
        const float g0 = dlds[(18 + 4 * a) * 33 + pl];
        const float g1 = dlds[(19 + 4 * a) * 33 + pl];
        const float g2 = dlds[(20 + 4 * a) * 33 + pl];
        const float g3 = dlds[(21 + 4 * a) * 33 + pl];
        float r0 = g0 + fy, r1 = g1 + fx;
        float r2 = g2 + g0 + fy, r3 = g3 + g1 + fx;
        r0 = fminf(fmaxf(r0, 0.f), 256.f);
        r1 = fminf(fmaxf(r1, 0.f), 256.f);
        r2 = fminf(fmaxf(r2, 0.f), 256.f);
        r3 = fminf(fmaxf(r3, 0.f), 256.f);
        float* o = score + ((size_t)p * AN + a) * 6;
        o[0] = c0; o[1] = c1; o[2] = r0; o[3] = r1; o[4] = r2; o[5] = r3;
    }
}

// =====================================================================
// Top-k: two-level radix histogram -> threshold -> collect -> bitonic sort.
// =====================================================================
__global__ void topk_init(int* ctl) {
    const int i = threadIdx.x;
    ctl[i] = 0;
    ctl[256 + i] = 0;
    if (i < 8) ctl[512 + i] = 0;
}

__global__ __launch_bounds__(256) void topk_hist1(const float* __restrict__ score, int* ctl) {
    __shared__ int lh[256];
    lh[threadIdx.x] = 0;
    __syncthreads();
    const size_t stride = (size_t)gridDim.x * 256;
    for (size_t i = (size_t)blockIdx.x * 256 + threadIdx.x; i < NSCORE; i += stride)
        atomicAdd(&lh[fkey(score[i * 6]) >> 24], 1);
    __syncthreads();
    atomicAdd(&ctl[threadIdx.x], lh[threadIdx.x]);
}

__global__ void topk_scan1(int* ctl) {
    if (threadIdx.x == 0) {
        int run = 0, b = 255;
        for (; b >= 0; --b) {
            const int h = ctl[b];
            if (run + h >= TOPK) break;
            run += h;
        }
        if (b < 0) b = 0;
        ctl[512] = b;
        ctl[513] = run;
    }
}

__global__ __launch_bounds__(256) void topk_hist2(const float* __restrict__ score, int* ctl) {
    __shared__ int lh[256];
    const unsigned int b1 = (unsigned int)ctl[512];
    lh[threadIdx.x] = 0;
    __syncthreads();
    const size_t stride = (size_t)gridDim.x * 256;
    for (size_t i = (size_t)blockIdx.x * 256 + threadIdx.x; i < NSCORE; i += stride) {
        const unsigned int k = fkey(score[i * 6]);
        if ((k >> 24) == b1) atomicAdd(&lh[(k >> 16) & 255u], 1);
    }
    __syncthreads();
    atomicAdd(&ctl[256 + threadIdx.x], lh[threadIdx.x]);
}

__global__ void topk_scan2(int* ctl) {
    if (threadIdx.x == 0) {
        const int b1 = ctl[512];
        const int need = TOPK - ctl[513];
        int run = 0, b2 = 255;
        for (; b2 >= 0; --b2) {
            const int h = ctl[256 + b2];
            if (run + h >= need) break;
            run += h;
        }
        if (b2 < 0) b2 = 0;
        ctl[514] = (int)(((unsigned)b1 << 24) | ((unsigned)b2 << 16));
    }
}

__global__ __launch_bounds__(256) void topk_collect(const float* __restrict__ score, int* ctl) {
    const unsigned int thr = (unsigned int)ctl[514];
    unsigned int* ck = (unsigned int*)(ctl + 1024);
    unsigned int* ci = ck + SORTN;
    const size_t stride = (size_t)gridDim.x * 256;
    for (size_t i = (size_t)blockIdx.x * 256 + threadIdx.x; i < NSCORE; i += stride) {
        const unsigned int k = fkey(score[i * 6]);
        if (k >= thr) {
            const int pos = atomicAdd(&ctl[515], 1);
            if (pos < CANDMAX) { ck[pos] = k; ci[pos] = (unsigned int)i; }
        }
    }
}

__global__ __launch_bounds__(1024) void topk_sort_gather(
    const float* __restrict__ score, const int* __restrict__ ctl, float* __restrict__ out)
{
    __shared__ unsigned long long s[SORTN]; // 64KB
    const unsigned int* ck = (const unsigned int*)(ctl + 1024);
    const unsigned int* ci = ck + SORTN;
    int cnt = ctl[515];
    if (cnt > CANDMAX) cnt = CANDMAX;

    for (int i = threadIdx.x; i < SORTN; i += 1024) {
        unsigned long long v = 0ull;
        if (i < cnt)
            v = ((unsigned long long)ck[i] << 32) | (unsigned long long)(~ci[i]);
        s[i] = v;
    }
    __syncthreads();

    for (int k = 2; k <= SORTN; k <<= 1) {
        for (int j = k >> 1; j > 0; j >>= 1) {
            for (int t = threadIdx.x; t < SORTN; t += 1024) {
                const int l = t ^ j;
                if (l > t) {
                    const unsigned long long a = s[t], b = s[l];
                    const bool descend = ((t & k) == 0);
                    if ((a < b) == descend) { s[t] = b; s[l] = a; }
                }
            }
            __syncthreads();
        }
    }

    for (int i = threadIdx.x; i < TOPK; i += 1024) {
        const unsigned long long v = s[i];
        float* dst = out + (size_t)i * 6;
        if (v == 0ull) {
            #pragma unroll
            for (int c = 0; c < 6; ++c) dst[c] = 0.f;
        } else {
            const unsigned int idx = ~((unsigned int)(v & 0xFFFFFFFFull));
            const float* src = score + (size_t)idx * 6;
            #pragma unroll
            for (int c = 0; c < 6; ++c) dst[c] = src[c];
        }
    }
}

// =====================================================================
extern "C" void kernel_launch(void* const* d_in, const int* in_sizes, int n_in,
                              void* d_out, int out_size, void* d_ws, size_t ws_size,
                              hipStream_t stream) {
    const float* x  = (const float*)d_in[0];
    const float* pw = (const float*)d_in[1];
    const float* pb = (const float*)d_in[2];
    const float* cw = (const float*)d_in[3];
    const float* cb = (const float*)d_in[4];
    const float* rw = (const float*)d_in[5];
    const float* rb = (const float*)d_in[6];
    float* out = (float*)d_out;

    char* ws = (char*)d_ws;
    unsigned short* feat = (unsigned short*)ws;                       // 67,108,864 B (bf16 [px][ch])
    float* score = (float*)(ws + 67108864);                           // 56,623,104 B
    int*   ctl   = (int*)(ws + 67108864 + 56623104);                  // control + candidates (69,632 B)
    unsigned short* wpk = (unsigned short*)(ws + 67108864 + 56623104 + 69632); // 589,824 B

    repack_weights<<<(9 * 256 * MED + 255) / 256, 256, 0, stream>>>(pw, wpk);
    topk_init<<<1, 256, 0, stream>>>(ctl);
    conv3x3_wmma<<<NPIX / 128, 256, 0, stream>>>(x, wpk, pb, feat);
    heads_decode_wmma<<<NPIX / 32, 256, 0, stream>>>(feat, cw, cb, rw, rb, score);
    topk_hist1<<<1024, 256, 0, stream>>>(score, ctl);
    topk_scan1<<<1, 32, 0, stream>>>(ctl);
    topk_hist2<<<1024, 256, 0, stream>>>(score, ctl);
    topk_scan2<<<1, 32, 0, stream>>>(ctl);
    topk_collect<<<1024, 256, 0, stream>>>(score, ctl);
    topk_sort_gather<<<1, 1024, 0, stream>>>(score, ctl, out);
    (void)in_sizes; (void)n_in; (void)out_size; (void)ws_size;
}
